// EncoderSimilarity_25744033972885
// MI455X (gfx1250) — matrix operand order repaired and checked by reading, not compiled, not run on hardware
//
#include <hip/hip_runtime.h>
#include <cstdint>

// MI455X / gfx1250, wave32. Fused EncoderSimilarity:
//   attn = softmax(9 * l2norm_q(leaky(ctx @ (query*matrix)^T)))      (WMMA bf16)
//   wc   = l2norm_d(attn @ ctx)                                      (WMMA bf16)
//   out  = l2norm_k((query - wc)^2 @ W^T + b)                        (WMMA bf16)

typedef __attribute__((ext_vector_type(16))) __bf16 v16bf;
typedef __attribute__((ext_vector_type(8)))  float  v8f;
typedef __attribute__((ext_vector_type(16))) unsigned short v16us;

#define DEV __device__ __forceinline__

static constexpr int Bb = 512, Qd = 64, Sd = 36, Dd = 1024, SIMd = 256;
static constexpr int SP = 48;   // S padded to 3 WMMA tiles for stage-1 N dim
static constexpr int SK = 64;   // S padded to 2 WMMA K-chunks for stage-2

DEV unsigned short f2bf(float f) {
  unsigned u = __builtin_bit_cast(unsigned, f);
  unsigned r = (u + 0x7FFFu + ((u >> 16) & 1u)) >> 16;   // round-to-nearest-even
  return (unsigned short)r;
}
DEV float bf2f(unsigned short s) {
  return __builtin_bit_cast(float, ((unsigned)s) << 16);
}

// A fragment: 16x32 bf16, row-major source with leading dim `ld` (elements).
// ISA 7.12.2: lanes 0-15 -> M=lane, lanes 16-31 -> M=lane-16;
// VGPR v holds K pair at (v&4?16:0) + (v&3)*2 + 8*(lane>>4).
DEV v16bf load_a_frag(const unsigned short* p, int ld, int k0, int lane) {
  const int m = lane & 15, h = lane >> 4;
  v16us u;
#pragma unroll
  for (int v = 0; v < 8; ++v) {
    const int kb = ((v & 4) ? 16 : 0) + (v & 3) * 2 + h * 8;
    const unsigned short* q = p + (size_t)m * ld + k0 + kb;
    u[2 * v]     = q[0];
    u[2 * v + 1] = q[1];
  }
  return __builtin_bit_cast(v16bf, u);
}

// B fragment: 32x16 bf16, row-major B[k][n] with leading dim `ld`.
// lanes 0-15 -> N=lane, lanes 16-31 -> N=lane-16; K order matches load_a_frag.
DEV v16bf load_b_frag(const unsigned short* p, int ld, int k0, int n0, int lane) {
  const int n = lane & 15, h = lane >> 4;
  v16us u;
#pragma unroll
  for (int j = 0; j < 16; ++j) {
    const int kk = ((j & 8) ? 16 : 0) + (j & 7) + h * 8;
    u[j] = p[(size_t)(k0 + kk) * ld + n0 + n];
  }
  return __builtin_bit_cast(v16bf, u);
}

DEV v8f wmma_bf16(v16bf a, v16bf b, v8f c) {
  // (neg_a, A, neg_b, B, c_mod, C, reuse_a, reuse_b)
  return __builtin_amdgcn_wmma_f32_16x16x32_bf16(false, a, false, b, (short)0, c,
                                                 false, false);
}

// ---------------------------------------------------------------- kernel 0
// WT[d][k] = bf16(W[k][d])   (SIM x D -> D x SIM), 1024 blocks x 256 thr.
__global__ void enc_sim_k0_wt(const float* __restrict__ W,
                              unsigned short* __restrict__ WT) {
  const int i = blockIdx.x * 256 + threadIdx.x;   // 0 .. 262143
  const int k = i & (SIMd - 1);
  const int d = i >> 8;
  WT[(size_t)d * SIMd + k] = f2bf(W[(size_t)k * Dd + d]);
}

// ---------------------------------------------------------------- kernel 1
// Per-batch attention weights -> attnW (B,Q,36) f32 in workspace.
// LDS: qm bf16 [64][1024] | ctxT bf16 [1024][48] | attn f32 [64][48] | cn[48]
static constexpr int K1_QM   = 0;
static constexpr int K1_CTXT = K1_QM + Qd * Dd * 2;              // 131072
static constexpr int K1_ATT  = K1_CTXT + Dd * SP * 2;            // 229376
static constexpr int K1_CN   = K1_ATT + Qd * SP * 4;             // 241664
static constexpr int K1_LDS  = K1_CN + SP * 4;                   // 241856

__global__ void enc_sim_k1_attn(const float* __restrict__ query,
                                const float* __restrict__ context,
                                const float* __restrict__ matrix,
                                const int* __restrict__ smooth_p,
                                float* __restrict__ attnW) {
  extern __shared__ char lds_raw[];
  unsigned short* qm   = (unsigned short*)(lds_raw + K1_QM);
  unsigned short* ctxT = (unsigned short*)(lds_raw + K1_CTXT);
  float*          attn = (float*)(lds_raw + K1_ATT);
  float*          cn   = (float*)(lds_raw + K1_CN);

  const int b = blockIdx.x, tid = threadIdx.x;
  const int lane = tid & 31, wave = tid >> 5;
  const float* qb = query  + (size_t)b * Qd * Dd;
  const float* mb = matrix + (size_t)b * Qd * Dd;
  const float* cb = context + (size_t)b * Sd * Dd;

  // qm = bf16(query * matrix), vectorized f32x4 loads
  for (int i = tid; i < Qd * Dd / 4; i += 256) {
    const float4 qv = ((const float4*)qb)[i];
    const float4 mv = ((const float4*)mb)[i];
    qm[4 * i + 0] = f2bf(qv.x * mv.x);
    qm[4 * i + 1] = f2bf(qv.y * mv.y);
    qm[4 * i + 2] = f2bf(qv.z * mv.z);
    qm[4 * i + 3] = f2bf(qv.w * mv.w);
  }
  // ctxT[d][s] = bf16(ctx[s][d]); s in [36,48) zero-padded
  for (int i = tid; i < Dd * SP; i += 256) {
    const int d = i / SP, s = i - d * SP;
    ctxT[i] = (s < Sd) ? f2bf(cb[(size_t)s * Dd + d]) : (unsigned short)0;
  }
  __syncthreads();

  // attn[q][s] = sum_d qm[q][d] * ctxT[d][s] ; fused LeakyReLU(0.1)
  for (int t = wave; t < 12; t += 8) {          // 4 q-tiles x 3 s-tiles
    const int qt = t & 3, st = t >> 2;
    v8f acc = {};
    for (int k0 = 0; k0 < Dd; k0 += 32) {
      const v16bf a  = load_a_frag(qm + (size_t)qt * 16 * Dd, Dd, k0, lane);
      const v16bf bb = load_b_frag(ctxT, SP, k0, st * 16, lane);
      acc = wmma_bf16(a, bb, acc);
    }
    const int n = lane & 15, h = lane >> 4;
#pragma unroll
    for (int v = 0; v < 8; ++v) {
      const float x = acc[v];
      attn[(qt * 16 + v + 8 * h) * SP + st * 16 + n] = (x > 0.f) ? x : 0.1f * x;
    }
  }
  __syncthreads();

  // l2norm over q for each s (padded s gives 0/eps = 0)
  if (tid < SP) {
    float s2 = 0.f;
    for (int q = 0; q < Qd; ++q) { const float x = attn[q * SP + tid]; s2 += x * x; }
    cn[tid] = 1.f / (sqrtf(s2) + 1e-8f);
  }
  __syncthreads();
  for (int i = tid; i < Qd * SP; i += 256) attn[i] *= cn[i % SP];
  __syncthreads();

  // temperature softmax over s, one q per thread; write to workspace
  const float sm = (float)(*smooth_p);
  if (tid < Qd) {
    const float* r = attn + tid * SP;
    float mx = -3.4e38f;
    for (int s = 0; s < Sd; ++s) mx = fmaxf(mx, r[s]);
    float sum = 0.f;
    for (int s = 0; s < Sd; ++s) sum += __expf((r[s] - mx) * sm);
    const float inv = 1.f / sum;
    float* ao = attnW + ((size_t)b * Qd + tid) * Sd;
    for (int s = 0; s < Sd; ++s) ao[s] = __expf((r[s] - mx) * sm) * inv;
  }
}

// ---------------------------------------------------------------- kernel 2
// Per-batch: wc = attn@ctx, l2norm; sim = (query-wc)^2; out = l2norm(sim@WT + b)
// LDS: [0)       ctx bf16 [64][1024]  -> reused as sim bf16
//      [131072)  wc  bf16 [64][1024]  -> reused as out_f f32 [64][256]
//      [262144)  at  bf16 [64][64]
//      [270336)  qn f32[64] ; [270592) rn f32[64]
static constexpr int K2_WC  = Qd * Dd * 2;                       // 131072
static constexpr int K2_AT  = K2_WC + Qd * Dd * 2;               // 262144
static constexpr int K2_QN  = K2_AT + Qd * SK * 2;               // 270336
static constexpr int K2_RN  = K2_QN + Qd * 4;                    // 270592
static constexpr int K2_LDS = K2_RN + Qd * 4;                    // 270848

__global__ void enc_sim_k2_fused(const float* __restrict__ query,
                                 const float* __restrict__ context,
                                 const float* __restrict__ bias,
                                 const float* __restrict__ attnW,
                                 const unsigned short* __restrict__ WT,
                                 float* __restrict__ out) {
  extern __shared__ char lds_raw[];
  unsigned short* ctx_sim = (unsigned short*)lds_raw;
  unsigned short* wc      = (unsigned short*)(lds_raw + K2_WC);
  float*          out_f   = (float*)(lds_raw + K2_WC);           // reuse
  unsigned short* at      = (unsigned short*)(lds_raw + K2_AT);
  float*          qn      = (float*)(lds_raw + K2_QN);
  float*          rn      = (float*)(lds_raw + K2_RN);

  const int b = blockIdx.x, tid = threadIdx.x;
  const int lane = tid & 31, wave = tid >> 5;
  const float* qb = query   + (size_t)b * Qd * Dd;
  const float* cb = context + (size_t)b * Sd * Dd;
  const float* aw = attnW   + (size_t)b * Qd * Sd;

  // context rows, s in [36,64) zero-padded (K padding for stage 2)
  for (int i = tid; i < SK * Dd; i += 256) {
    const int r = i >> 10;
    ctx_sim[i] = (r < Sd) ? f2bf(cb[i]) : (unsigned short)0;
  }
  // attention weights, cols [36,64) zero-padded
  for (int i = tid; i < Qd * SK; i += 256) {
    const int q = i >> 6, s = i & 63;
    at[i] = (s < Sd) ? f2bf(aw[q * Sd + s]) : (unsigned short)0;
  }
  __syncthreads();

  // stage 2: wc[q][d] = sum_s at[q][s] * ctx[s][d]   (K = 64 = 2 chunks)
  for (int t = wave; t < 256; t += 8) {         // 4 q-tiles x 64 d-tiles
    const int qt = t & 3, dt = t >> 2;
    v8f acc = {};
    v16bf a0 = load_a_frag(at + (size_t)qt * 16 * SK, SK, 0, lane);
    v16bf b0 = load_b_frag(ctx_sim, Dd, 0, dt * 16, lane);
    acc = wmma_bf16(a0, b0, acc);
    v16bf a1 = load_a_frag(at + (size_t)qt * 16 * SK, SK, 32, lane);
    v16bf b1 = load_b_frag(ctx_sim, Dd, 32, dt * 16, lane);
    acc = wmma_bf16(a1, b1, acc);
    const int n = lane & 15, h = lane >> 4;
#pragma unroll
    for (int v = 0; v < 8; ++v)
      wc[(size_t)(qt * 16 + v + 8 * h) * Dd + dt * 16 + n] = f2bf(acc[v]);
  }
  __syncthreads();

  // l2norm of wc over d
  if (tid < Qd) {
    float s2 = 0.f;
    const unsigned short* r = wc + (size_t)tid * Dd;
    for (int d = 0; d < Dd; ++d) { const float x = bf2f(r[d]); s2 += x * x; }
    qn[tid] = 1.f / (sqrtf(s2) + 1e-8f);
  }
  __syncthreads();

  // sim = (query - wc_hat)^2, overwriting the ctx region
  for (int i = tid; i < Qd * Dd; i += 256) {
    const int q = i >> 10;
    const float w = bf2f(wc[i]) * qn[q];
    const float d = qb[i] - w;
    ctx_sim[i] = f2bf(d * d);
  }
  __syncthreads();

  // stage 3: out[q][k] = sum_d sim[q][d] * WT[d][k] + bias[k]
  {
    const int qt = wave & 3;
    const int nb = (wave >> 2) * 8;             // 8 n-tiles per wave
    v8f acc[8];
    const v8f zero = {};
#pragma unroll
    for (int j = 0; j < 8; ++j) acc[j] = zero;
    for (int k0 = 0; k0 < Dd; k0 += 32) {
      const v16bf a = load_a_frag(ctx_sim + (size_t)qt * 16 * Dd, Dd, k0, lane);
#pragma unroll
      for (int j = 0; j < 8; ++j) {
        const v16bf bb = load_b_frag(WT, SIMd, k0, (nb + j) * 16, lane);
        acc[j] = wmma_bf16(a, bb, acc[j]);
      }
    }
    const int n = lane & 15, h = lane >> 4;
#pragma unroll
    for (int j = 0; j < 8; ++j) {
      const int col = (nb + j) * 16 + n;
      const float bv = bias[col];
#pragma unroll
      for (int v = 0; v < 8; ++v)
        out_f[(qt * 16 + v + 8 * h) * SIMd + col] = acc[j][v] + bv;
    }
  }
  __syncthreads();

  // final l2norm over SIM, write output
  if (tid < Qd) {
    float s2 = 0.f;
    const float* r = out_f + tid * SIMd;
    for (int k = 0; k < SIMd; ++k) s2 += r[k] * r[k];
    rn[tid] = 1.f / (sqrtf(s2) + 1e-8f);
  }
  __syncthreads();
  float* ob = out + (size_t)b * Qd * SIMd;
  for (int i = tid; i < Qd * SIMd; i += 256) ob[i] = out_f[i] * rn[i >> 8];
}

// ---------------------------------------------------------------- launch
extern "C" void kernel_launch(void* const* d_in, const int* in_sizes, int n_in,
                              void* d_out, int out_size, void* d_ws, size_t ws_size,
                              hipStream_t stream) {
  (void)in_sizes; (void)n_in; (void)out_size; (void)ws_size;
  const float* query   = (const float*)d_in[0];
  const float* context = (const float*)d_in[1];
  const float* matrix  = (const float*)d_in[2];
  const float* W       = (const float*)d_in[3];
  const float* bias    = (const float*)d_in[4];
  const int*   smooth  = (const int*)d_in[5];
  float* out = (float*)d_out;

  // workspace: WT bf16 (D x SIM) = 512 KB, then attn weights f32 (B,Q,36) ~4.7 MB
  unsigned short* WT = (unsigned short*)d_ws;
  float* attnW = (float*)((char*)d_ws + (size_t)Dd * SIMd * 2);

  enc_sim_k0_wt<<<(Dd * SIMd) / 256, 256, 0, stream>>>(W, WT);
  enc_sim_k1_attn<<<Bb, 256, K1_LDS, stream>>>(query, context, matrix, smooth, attnW);
  enc_sim_k2_fused<<<Bb, 256, K2_LDS, stream>>>(query, context, bias, attnW, WT, out);
}
